// Score_dur_to_note_dur_317827580763
// MI455X (gfx1250) — matrix operand order repaired and checked by reading, not compiled, not run on hardware
//
#include <hip/hip_runtime.h>
#include <hip/hip_bf16.h>
#include <stdint.h>
#include <stddef.h>

// Problem constants (from reference)
#define B_   32
#define P_   4096
#define N_   1024
#define D_   256
#define POS_ 10
#define H_   256
#define G4H  1024     // 4*H
#define CP0  288      // D+POS=266 padded to mult of 32
#define CPE  288      // LSTM0 input 258 padded

typedef __attribute__((ext_vector_type(16))) __bf16 v16bf;
typedef __attribute__((ext_vector_type(8)))  float  v8f;
typedef __attribute__((ext_vector_type(4)))  unsigned int u32x4;

union FragAB { u32x4 u[2]; v16bf v; };

// A-matrix fragment (16x32 bf16, M x K), src row-major [row][k], ldk elems.
__device__ __forceinline__ v16bf ldA(const __bf16* base, int ldk, int row, int k0) {
  FragAB f;
  const __bf16* p = base + (size_t)row * ldk + k0 + (k0 & 8);
  f.u[0] = *reinterpret_cast<const u32x4*>(p);
  f.u[1] = *reinterpret_cast<const u32x4*>(p + 8);
  return f.v;
}
// B-matrix fragment (32x16 bf16, K x N), column n = lane&15 -> storage row.
__device__ __forceinline__ v16bf ldB(const __bf16* base, int ldk, int row, int k0) {
  FragAB f;
  const __bf16* p = base + (size_t)row * ldk + k0;
  f.u[0] = *reinterpret_cast<const u32x4*>(p);
  f.u[1] = *reinterpret_cast<const u32x4*>(p + 8);
  return f.v;
}
__device__ __forceinline__ v8f wmma_bf16(v16bf a, v16bf b, v8f c) {
  return __builtin_amdgcn_wmma_f32_16x16x32_bf16(false, a, false, b, (short)0, c, false, false);
}
__device__ __forceinline__ float sigf(float x)     { return 1.f / (1.f + __expf(-x)); }
__device__ __forceinline__ float tanhfast(float x) { return 2.f * sigf(2.f * x) - 1.f; }

// ---------------- weight packing (fp32 -> padded bf16) ----------------
__global__ void pack_conv_w(const float* __restrict__ w, __bf16* __restrict__ out,
                            int cout, int cin, int cpad) {
  int total = 3 * cout * cpad;
  for (int i = blockIdx.x * blockDim.x + threadIdx.x; i < total; i += gridDim.x * blockDim.x) {
    int k = i % cpad; int tmp = i / cpad; int co = tmp % cout; int dt = tmp / cout;
    float v = (k < cin) ? w[((size_t)co * cin + k) * 3 + dt] : 0.f;
    out[i] = (__bf16)v;   // layout [dt][co][cpad]
  }
}
__global__ void pack_mat_w(const float* __restrict__ w, __bf16* __restrict__ out,
                           int rows, int kin, int kpad) {
  int total = rows * kpad;
  for (int i = blockIdx.x * blockDim.x + threadIdx.x; i < total; i += gridDim.x * blockDim.x) {
    int k = i % kpad; int r = i / kpad;
    float v = (k < kin) ? w[(size_t)r * kin + k] : 0.f;
    out[i] = (__bf16)v;   // layout [rows][kpad]
  }
}

// ---------------- embedding + positional concat ----------------
__global__ void embed_concat(const int* __restrict__ seq, const int* __restrict__ order,
                             const float* __restrict__ emb, const float* __restrict__ pos,
                             __bf16* __restrict__ x0) {
  int total = B_ * P_ * CP0;
  for (int i = blockIdx.x * blockDim.x + threadIdx.x; i < total; i += gridDim.x * blockDim.x) {
    int c = i % CP0; int t = (i / CP0) % P_; int b = i / (CP0 * P_);
    float v = 0.f;
    if (c < D_)             v = emb[seq[b * P_ + t] * D_ + c];
    else if (c < D_ + POS_) v = pos[order[b * P_ + t] * POS_ + (c - D_)];
    x0[i] = (__bf16)v;
  }
}

// ---------------- k=3 conv1d as shifted WMMA GEMM ----------------
// in  : bf16 [B][L][CPAD]   (channel-contiguous)
// wp  : bf16 [3][256][CPAD] (tap-major, row-major M x K)
// out : bf16 [B][L][256]
template <int CPAD, bool RELU>
__global__ void conv3_gemm(const __bf16* __restrict__ xin, const __bf16* __restrict__ wp,
                           const float* __restrict__ bias, __bf16* __restrict__ out, int L) {
  __shared__ __bf16 sx[18 * CPAD];       // time window t0-1 .. t0+16
  int ntT = L >> 4;
  int grp = blockIdx.x & 1;              // 2 groups x 8 waves = 16 cout tiles
  int tmp = blockIdx.x >> 1;
  int tt  = tmp % ntT;
  int b   = tmp / ntT;
  int t0  = tt << 4;
  const __bf16* xb = xin + (size_t)b * L * CPAD;
  for (int i = threadIdx.x; i < 18 * CPAD; i += blockDim.x) {
    int row = i / CPAD, k = i - row * CPAD;
    int t = t0 + row - 1;
    __bf16 v = (__bf16)0.f;
    if (t >= 0 && t < L) v = xb[(size_t)t * CPAD + k];
    sx[i] = v;
  }
  __syncthreads();
  int lane = threadIdx.x & 31;
  int wv   = threadIdx.x >> 5;
  int coBase = (grp * 8 + wv) << 4;
  int half = lane >> 4, l15 = lane & 15;
  v8f acc = {};
#pragma unroll
  for (int dt = 0; dt < 3; ++dt) {
    const __bf16* wtap = wp + (size_t)dt * 256 * CPAD;
#pragma unroll
    for (int kk = 0; kk < CPAD / 32; ++kk) {
      v16bf a  = ldA(wtap, CPAD, coBase + l15, kk * 32 + (half ? 8 : 0));
      v16bf bf = ldB(sx,   CPAD, l15 + dt,     kk * 32 + (half ? 16 : 0));
      acc = wmma_bf16(a, bf, acc);
    }
  }
  int t = t0 + l15;
  int mOff = half ? 8 : 0;
#pragma unroll
  for (int r = 0; r < 8; ++r) {
    int co = coBase + mOff + r;
    float v = acc[r] + bias[co];
    if (RELU) v = v > 0.f ? v : 0.f;
    out[((size_t)b * L + t) * 256 + co] = (__bf16)v;
  }
}

// ---------------- note-segment ids (run-length scan, per batch) ----------------
__global__ void segment_ids(const int* __restrict__ seq, int* __restrict__ seg) {
  int b = blockIdx.x;
  const int* sb = seq + b * P_;
  __shared__ int ps[256];
  int tid = threadIdx.x;
  int base = tid * 16;
  int cnt = 0;
#pragma unroll
  for (int i = 0; i < 16; ++i) {
    int t = base + i;
    bool v  = (sb[t] > 1) && (t != P_ - 1);
    bool pv = (t > 0) && (sb[t - 1] > 1);
    cnt += (v && !pv) ? 1 : 0;
  }
  ps[tid] = cnt;
  __syncthreads();
  for (int off = 1; off < 256; off <<= 1) {
    int x = (tid >= off) ? ps[tid - off] : 0;
    __syncthreads();
    ps[tid] += x;
    __syncthreads();
  }
  int run = ps[tid] - cnt;   // exclusive start-count
  for (int i = 0; i < 16; ++i) {
    int t = base + i;
    bool v  = (sb[t] > 1) && (t != P_ - 1);
    bool pv = (t > 0) && (sb[t - 1] > 1);
    if (v && !pv) run++;
    int s = v ? (run - 1) : N_;
    if (s > N_) s = N_;          // overflow -> dump segment
    seg[b * P_ + t] = s;
  }
}

__global__ void zero_f32(float* p, int n) {
  for (int i = blockIdx.x * blockDim.x + threadIdx.x; i < n; i += gridDim.x * blockDim.x)
    p[i] = 0.f;
}

__global__ void seg_accum(const __bf16* __restrict__ x, const int* __restrict__ seg,
                          float* __restrict__ agg, float* __restrict__ cnt) {
  int bt = blockIdx.x;                 // b*P + t
  int s = seg[bt];
  if (s >= N_) return;
  int b = bt >> 12;                    // / 4096
  float* dst = agg + (size_t)(b * N_ + s) * 256;
  const __bf16* src = x + (size_t)bt * 256;
  atomicAdd(dst + threadIdx.x, (float)src[threadIdx.x]);
  if (threadIdx.x == 0) atomicAdd(cnt + b * N_ + s, 1.f);
}

__global__ void build_enc(const float* __restrict__ agg, const float* __restrict__ cnt,
                          const float* __restrict__ dur, __bf16* __restrict__ enc) {
  int total = B_ * N_ * CPE;
  for (int i = blockIdx.x * blockDim.x + threadIdx.x; i < total; i += gridDim.x * blockDim.x) {
    int c = i % CPE; int n = (i / CPE) % N_; int b = i / (CPE * N_);
    float v = 0.f;
    if (c < 256) {
      float ct = cnt[b * N_ + n];
      v = agg[(size_t)(b * N_ + n) * 256 + c] / fmaxf(ct, 1.f);
    } else if (c == 256) v = dur[b * N_ + n];
    else if (c == 257)   v = 1.f / (dur[b * N_ + n] + 1.f);
    enc[i] = (__bf16)v;
  }
}

// ---------------- LSTM input projection: xproj[dir][t][b][4H] = act @ Wih^T ----------------
template <int KPAD>
__global__ void proj_gemm(const __bf16* __restrict__ act,   // [B][N_][KPAD]
                          const __bf16* __restrict__ wih,   // [2][4H][KPAD]
                          float* __restrict__ xproj) {      // [2][N_][B][4H]
  int dir = blockIdx.y;
  int grp = blockIdx.x & 7;
  int pt  = blockIdx.x >> 3;
  int p0  = pt << 4;
  int b   = p0 >> 10;
  int t0  = p0 & 1023;
  const __bf16* wd = wih + (size_t)dir * G4H * KPAD;
  const __bf16* ab = act + (size_t)b * N_ * KPAD;
  float* xp = xproj + (size_t)dir * N_ * B_ * G4H;
  int lane = threadIdx.x & 31, wv = threadIdx.x >> 5;
  int gBase = (grp * 8 + wv) << 4;
  int half = lane >> 4, l15 = lane & 15;
  v8f acc = {};
#pragma unroll
  for (int kk = 0; kk < KPAD / 32; ++kk) {
    v16bf a  = ldA(wd, KPAD, gBase + l15, kk * 32 + (half ? 8 : 0));
    v16bf bf = ldB(ab, KPAD, t0 + l15,    kk * 32 + (half ? 16 : 0));
    acc = wmma_bf16(a, bf, acc);
  }
  int t = t0 + l15, mOff = half ? 8 : 0;
#pragma unroll
  for (int r = 0; r < 8; ++r)
    xp[((size_t)t * B_ + b) * G4H + gBase + mOff + r] = acc[r];
}

// ---------------- persistent recurrent step kernel (one block per direction) ----------------
// z_t = xproj[t] + h_{t-1} @ Whh^T ; gates i,f,g,o ; y[b][t][dir*256+j] = h_t
// LDS: Z[32][1024] f32, C[32][256] f32, Hs[32][256] bf16, BS[1024] f32 (bias sum)
__global__ void lstm_step(const float* __restrict__ xproj,  // [2][N_][B][4H]
                          const __bf16* __restrict__ whh,   // [2][4H][H]
                          const float* __restrict__ bih, const float* __restrict__ bhh,
                          __bf16* __restrict__ yout) {      // [B][N_][512]
  extern __shared__ char smem[];
  float*  Z  = (float*)smem;                                   // 128 KB
  float*  C  = (float*)(smem + 32 * 1024 * 4);                 //  32 KB
  __bf16* Hs = (__bf16*)(smem + 32 * 1024 * 4 + 32 * 256 * 4); //  16 KB
  float*  BS = (float*)(smem + 32 * 1024 * 4 + 32 * 256 * 4 + 32 * 256 * 2); // 4 KB
  int dir = blockIdx.x;
  const float*  xp = xproj + (size_t)dir * N_ * B_ * G4H;
  const __bf16* wd = whh + (size_t)dir * G4H * H_;
  int tid = threadIdx.x, lane = tid & 31, wv = tid >> 5;
  for (int i = tid; i < 32 * 256; i += 1024) { C[i] = 0.f; Hs[i] = (__bf16)0.f; }
  if (tid < G4H) BS[tid] = bih[dir * G4H + tid] + bhh[dir * G4H + tid];
  __syncthreads();
  int half = lane >> 4, l15 = lane & 15, mOff = half ? 8 : 0;
  for (int s = 0; s < N_; ++s) {
    int t = dir ? (N_ - 1 - s) : s;
    const float* zp = xp + (size_t)t * B_ * G4H;
    // prefetch next timestep's xproj slab while this step computes
    if (s + 1 < N_) {
      int tn = dir ? (N_ - 2 - s) : (s + 1);
      __builtin_prefetch(xp + (size_t)tn * B_ * G4H + tid * 32, 0, 1);
    }
    // Launder an *integer offset* (not the pointer!) so:
    //  - pointer provenance survives -> loads stay global_load_b128;
    //  - the offset is opaque and redefined per iteration, so LLVM cannot
    //    hoist the loop-invariant Whh fragment loads out of the time loop
    //    (which would blow the VGPR budget and spill to scratch).
    //  Whh is L2-resident (512 KB/dir); re-reading each step is cheap.
    unsigned woff = 0;
    asm volatile("" : "+v"(woff));
    const __bf16* wds = wd + woff;
    // Per wave: 2 gate tiles (nt = wv, wv+32) x 2 batch tiles (mt = 0,1).
    // Load each gate tile's 8 Whh B-fragments ONCE (one clause, 64 VGPRs),
    // then run both batch tiles (16 WMMAs) against them: halves Whh traffic
    // and gives each load burst ~16 WMMAs of latency cover.
#pragma unroll 1
    for (int g = 0; g < 2; ++g) {
      int nBase = (wv + g * 32) << 4;
      v16bf bfr[8];
#pragma unroll
      for (int kk = 0; kk < 8; ++kk)
        bfr[kk] = ldB(wds, 256, nBase + l15, kk * 32 + (half ? 16 : 0));
#pragma unroll 1
      for (int m = 0; m < 2; ++m) {
        int mBase = m << 4;
        // xproj values: issued now, consumed only at the epilogue, so the
        // 8-WMMA chain hides their latency.
        v8f zval;
#pragma unroll
        for (int r = 0; r < 8; ++r)
          zval[r] = zp[(mBase + mOff + r) * G4H + nBase + l15];
        v8f acc = {};
#pragma unroll
        for (int kk = 0; kk < 8; ++kk) {
          v16bf a = ldA(Hs, 256, mBase + l15, kk * 32 + (half ? 8 : 0));  // LDS
          acc = wmma_bf16(a, bfr[kk], acc);
        }
#pragma unroll
        for (int r = 0; r < 8; ++r)
          Z[(mBase + mOff + r) * G4H + nBase + l15] = acc[r] + zval[r];
      }
    }
    __syncthreads();
    for (int q = tid; q < 32 * 256; q += 1024) {
      int bb = q >> 8, j = q & 255;
      float zi = Z[bb * G4H + j]       + BS[j];
      float zf = Z[bb * G4H + 256 + j] + BS[256 + j];
      float zg = Z[bb * G4H + 512 + j] + BS[512 + j];
      float zo = Z[bb * G4H + 768 + j] + BS[768 + j];
      float c = sigf(zf) * C[q] + sigf(zi) * tanhfast(zg);
      C[q] = c;
      float hv = sigf(zo) * tanhfast(c);
      Hs[q] = (__bf16)hv;
      yout[((size_t)bb * N_ + t) * 512 + dir * 256 + j] = (__bf16)hv;
    }
    __syncthreads();
  }
}

// ---------------- final pointwise conv (Cout=1, k=3) ----------------
__global__ void cnn_final(const __bf16* __restrict__ h, const float* __restrict__ w,
                          const float* __restrict__ bias, float* __restrict__ out) {
  int idx = blockIdx.x * blockDim.x + threadIdx.x;   // b*N + t
  if (idx >= B_ * N_) return;
  int b = idx >> 10, t = idx & 1023;
  float acc = bias[0];
  for (int dt = 0; dt < 3; ++dt) {
    int tt = t + dt - 1;
    if (tt < 0 || tt >= N_) continue;
    const __bf16* hp = h + (size_t)(b * N_ + tt) * 256;
    for (int c = 0; c < 256; ++c) acc += w[c * 3 + dt] * (float)hp[c];
  }
  out[idx] = acc;
}

// ======================================================================
extern "C" void kernel_launch(void* const* d_in, const int* in_sizes, int n_in,
                              void* d_out, int out_size, void* d_ws, size_t ws_size,
                              hipStream_t stream) {
  const float* dur    = (const float*)d_in[0];
  const int*   seq    = (const int*)  d_in[1];
  const int*   order  = (const int*)  d_in[2];
  const float* emb    = (const float*)d_in[3];
  const float* posemb = (const float*)d_in[4];
  const float* mw1 = (const float*)d_in[5];  const float* mb1 = (const float*)d_in[6];
  const float* mw2 = (const float*)d_in[7];  const float* mb2 = (const float*)d_in[8];
  const float* wih0 = (const float*)d_in[9];  const float* whh0 = (const float*)d_in[10];
  const float* bih0 = (const float*)d_in[11]; const float* bhh0 = (const float*)d_in[12];
  const float* wih1 = (const float*)d_in[13]; const float* whh1 = (const float*)d_in[14];
  const float* bih1 = (const float*)d_in[15]; const float* bhh1 = (const float*)d_in[16];
  const float* cw1 = (const float*)d_in[17]; const float* cb1 = (const float*)d_in[18];
  const float* cw2 = (const float*)d_in[19]; const float* cb2 = (const float*)d_in[20];
  float* outp = (float*)d_out;

  char* ws = (char*)d_ws;
  size_t o = 0;
  auto alloc = [&](size_t bytes) { o = (o + 255) & ~(size_t)255; size_t r = o; o += bytes; return r; };
  __bf16* X0   = (__bf16*)(ws + alloc((size_t)B_ * P_ * CP0 * 2));
  __bf16* X1   = (__bf16*)(ws + alloc((size_t)B_ * P_ * 256 * 2));
  __bf16* X2   = (__bf16*)(ws + alloc((size_t)B_ * P_ * 256 * 2));
  int*    SEG  = (int*)   (ws + alloc((size_t)B_ * P_ * 4));
  float*  AGG  = (float*) (ws + alloc((size_t)B_ * N_ * 256 * 4));
  float*  CNT  = (float*) (ws + alloc((size_t)B_ * N_ * 4));
  __bf16* ENC  = (__bf16*)(ws + alloc((size_t)B_ * N_ * CPE * 2));
  float*  XPRJ = (float*) (ws + alloc((size_t)2 * N_ * B_ * G4H * 4));
  __bf16* Y0   = (__bf16*)(ws + alloc((size_t)B_ * N_ * 512 * 2));
  __bf16* Y1   = (__bf16*)(ws + alloc((size_t)B_ * N_ * 512 * 2));
  __bf16* C1   = (__bf16*)(ws + alloc((size_t)B_ * N_ * 256 * 2));
  __bf16* WM1  = (__bf16*)(ws + alloc((size_t)3 * 256 * CP0 * 2));
  __bf16* WM2  = (__bf16*)(ws + alloc((size_t)3 * 256 * 256 * 2));
  __bf16* WI0  = (__bf16*)(ws + alloc((size_t)2 * G4H * CPE * 2));
  __bf16* WH0  = (__bf16*)(ws + alloc((size_t)2 * G4H * 256 * 2));
  __bf16* WI1  = (__bf16*)(ws + alloc((size_t)2 * G4H * 512 * 2));
  __bf16* WH1  = (__bf16*)(ws + alloc((size_t)2 * G4H * 256 * 2));
  __bf16* WC1  = (__bf16*)(ws + alloc((size_t)3 * 256 * 512 * 2));

  const size_t LSTM_LDS = 32 * 1024 * 4 + 32 * 256 * 4 + 32 * 256 * 2 + G4H * 4;

  // 1) pack weights to padded bf16
  pack_conv_w<<<256, 256, 0, stream>>>(mw1, WM1, 256, 266, CP0);
  pack_conv_w<<<256, 256, 0, stream>>>(mw2, WM2, 256, 256, 256);
  pack_conv_w<<<256, 256, 0, stream>>>(cw1, WC1, 256, 512, 512);
  pack_mat_w <<<256, 256, 0, stream>>>(wih0, WI0, 2 * G4H, 258, CPE);
  pack_mat_w <<<256, 256, 0, stream>>>(whh0, WH0, 2 * G4H, 256, 256);
  pack_mat_w <<<256, 256, 0, stream>>>(wih1, WI1, 2 * G4H, 512, 512);
  pack_mat_w <<<256, 256, 0, stream>>>(whh1, WH1, 2 * G4H, 256, 256);

  // 2) embedding + positional concat -> X0 [B][P][288] bf16
  embed_concat<<<4096, 256, 0, stream>>>(seq, order, emb, posemb, X0);

  // 3) mix convs (WMMA)
  conv3_gemm<CP0, true ><<<B_ * (P_ / 16) * 2, 256, 0, stream>>>(X0, WM1, mb1, X1, P_);
  conv3_gemm<256, false><<<B_ * (P_ / 16) * 2, 256, 0, stream>>>(X1, WM2, mb2, X2, P_);

  // 4) note-boundary segmentation + mean pooling
  segment_ids<<<B_, 256, 0, stream>>>(seq, SEG);
  zero_f32<<<2048, 256, 0, stream>>>(AGG, B_ * N_ * 256);
  zero_f32<<<64,   256, 0, stream>>>(CNT, B_ * N_);
  seg_accum<<<B_ * P_, 256, 0, stream>>>(X2, SEG, AGG, CNT);
  build_enc<<<4096, 256, 0, stream>>>(AGG, CNT, dur, ENC);

  // 5) BiLSTM layer 0: input projection (WMMA) + persistent recurrence (WMMA)
  proj_gemm<CPE><<<dim3((B_ * N_ / 16) * 8, 2), 256, 0, stream>>>(ENC, WI0, XPRJ);
  lstm_step<<<2, 1024, LSTM_LDS, stream>>>(XPRJ, WH0, bih0, bhh0, Y0);

  // 6) BiLSTM layer 1
  proj_gemm<512><<<dim3((B_ * N_ / 16) * 8, 2), 256, 0, stream>>>(Y0, WI1, XPRJ);
  lstm_step<<<2, 1024, LSTM_LDS, stream>>>(XPRJ, WH1, bih1, bhh1, Y1);

  // 7) head convs
  conv3_gemm<512, true><<<B_ * (N_ / 16) * 2, 256, 0, stream>>>(Y1, WC1, cb1, C1, N_);
  cnn_final<<<(B_ * N_ + 255) / 256, 256, 0, stream>>>(C1, cw2, cb2, outp);
}
